// Discriminator_87170656239864
// MI455X (gfx1250) — matrix operand order repaired and checked by reading, not compiled, not run on hardware
//
#include <hip/hip_runtime.h>
#include <hip/hip_bf16.h>

// ---------------------------------------------------------------------------
// Types for CDNA5 WMMA (wave32): v_wmma_f32_16x16x32_bf16
// ---------------------------------------------------------------------------
typedef __attribute__((ext_vector_type(16))) __bf16 v16bf;
typedef __attribute__((ext_vector_type(8)))  __bf16 v8bf;
typedef __attribute__((ext_vector_type(4)))  __bf16 v4bf;
typedef __attribute__((ext_vector_type(8)))  float  v8f;
typedef __attribute__((ext_vector_type(4)))  float  v4f;

__device__ __forceinline__ __bf16 f2bf(float f) { return (__bf16)f; }
__device__ __forceinline__ float  bf2f(__bf16 b) { return (float)b; }
__device__ __forceinline__ float  sigf(float x) { return 1.f / (1.f + __expf(-x)); }

__device__ __forceinline__ v8f wmma_bf16(v16bf a, v16bf b, v8f c) {
    // (neg_a, A, neg_b, B, c_mod, C, reuse_a, reuse_b)
    return __builtin_amdgcn_wmma_f32_16x16x32_bf16(false, a, false, b, (short)0, c, false, false);
}

// A-matrix 16x32 bf16 fragment. Tile is ROW-major [16][lda]. Per ISA 7.12.2,
// lane m, half h holds K = {8h..8h+7} and {16+8h..16+8h+7}: two b128 loads.
__device__ __forceinline__ v16bf frag_A(const __bf16* p, int lda, int lane) {
    const __bf16* q = p + (lane & 15) * lda + (lane >> 4) * 8;
    v8bf lo = *(const v8bf*)(q);
    v8bf hi = *(const v8bf*)(q + 16);
    return __builtin_shufflevector(lo, hi, 0,1,2,3,4,5,6,7,8,9,10,11,12,13,14,15);
}
// B-matrix 32x16 bf16 fragment. Tile is COLUMN-major [n][ldb] (each row = one
// output column, 32 K values). lane n, half h holds K = {8h..8h+7} and {16+8h..16+8h+7}: two b128.
__device__ __forceinline__ v16bf frag_B(const __bf16* p, int ldb, int lane) {
    const __bf16* q = p + (lane & 15) * ldb + (lane >> 4) * 8;
    v8bf lo = *(const v8bf*)(q);
    v8bf hi = *(const v8bf*)(q + 16);
    return __builtin_shufflevector(lo, hi, 0,1,2,3,4,5,6,7,8,9,10,11,12,13,14,15);
}

// ---------------------------------------------------------------------------
// Problem constants
// ---------------------------------------------------------------------------
#define BB   512
#define RW   16
#define NN   12000
#define HI   128
#define HH   512
#define HT   12
#define ROWS (BB * RW)      // 8192
#define KPAD 160            // HI + HT padded to multiple of 32

// ---------------------------------------------------------------------------
// Weight prep: f32 [J][K] row-major  ->  bf16 panels [Kpad/32][J][32]
// (zero-pad K). J must be a power of two (lj = log2 J).
// Panel layout makes the global->LDS B-tile copy contiguous per column.
// ---------------------------------------------------------------------------
__global__ void k_panel_cvt(const float* __restrict__ src, __bf16* __restrict__ dst,
                            int lj, int K, int Kpad) {
    int idx = blockIdx.x * blockDim.x + threadIdx.x;
    int J = 1 << lj;
    if (idx >= Kpad * J) return;
    int kk = idx & 31;
    int j  = (idx >> 5) & (J - 1);
    int p  = idx >> (5 + lj);
    int k  = p * 32 + kk;
    float val = (k < K) ? src[(size_t)j * K + k] : 0.f;
    dst[idx] = f2bf(val);
}

__global__ void k_init_state(__bf16* __restrict__ h0, float* __restrict__ c, int n) {
    int idx = blockIdx.x * blockDim.x + threadIdx.x;
    if (idx < n) { h0[idx] = f2bf(0.f); c[idx] = 0.f; }
}

// ---------------------------------------------------------------------------
// Fused: v_d = v @ W_down^T ; vt = [v_d | t*w+b | 0pad] ; inputs = vt @ W_vt^T
// Block: 32 rows x full HI=128. 8 waves; wave w -> mt = w&1, nt = {2(w>>1), +1}
// ---------------------------------------------------------------------------
__global__ __launch_bounds__(256) void k_downproj_fused(
    const float* __restrict__ v,      // [8192, 12000]
    const float* __restrict__ t,      // [8192]
    const float* __restrict__ wtw,    // [12]
    const float* __restrict__ wtb,    // [12]
    const __bf16* __restrict__ WdP,   // panels [375][128][32]
    const __bf16* __restrict__ WvtP,  // panels [5][128][32]
    __bf16* __restrict__ inputs)      // [8192, 128]
{
    __shared__ __align__(16) __bf16 As[32][56];     // row-major A tile
    __shared__ __align__(16) __bf16 BsT[128][40];   // column-major B tile
    __shared__ __align__(16) __bf16 VT[32][168];    // vt tile (row-major A)
    __shared__ __align__(16) __bf16 Wv[128][168];   // W_vt^T column-major [n][k]

    const int tid  = threadIdx.x;
    const int lane = tid & 31;
    const int w    = tid >> 5;
    const int mt   = w & 1;
    const int nt0  = (w >> 1) * 2;
    const int rb   = blockIdx.x * 32;

    // preload W_vt panels into column-major LDS tile (uint4-wide)
#pragma unroll
    for (int e = 0; e < 10; ++e) {
        int idx = tid + e * 256;            // 2560 uint4 = 128 cols * 160 k
        int p = idx >> 9, r = idx & 511;
        int n = r >> 2, q = r & 3;
        *(uint4*)&Wv[n][p * 32 + q * 8] =
            *(const uint4*)(WvtP + ((size_t)(p * 128 + n)) * 32 + q * 8);
    }

    v8f acc0 = {}, acc1 = {};
    for (int k0 = 0; k0 < NN; k0 += 32) {
        {   // A: 32x32 f32 -> bf16: v4f load, packed cvt, b64 LDS store
            int m = tid >> 3, q = tid & 7;
            v4f f = *(const v4f*)(v + (size_t)(rb + m) * NN + k0 + q * 4);
            *(v4bf*)&As[m][q * 4] = __builtin_convertvector(f, v4bf);
        }
        {   // B: copy panel (128 cols x 32 k contiguous) -> column-major LDS
            const __bf16* panel = WdP + (size_t)(k0 >> 5) * (128 * 32);
#pragma unroll
            for (int e = 0; e < 2; ++e) {
                int idx = tid + e * 256;    // 512 uint4
                int n = idx >> 2, q = idx & 3;
                *(uint4*)&BsT[n][q * 8] = *(const uint4*)(panel + (size_t)n * 32 + q * 8);
            }
        }
        __syncthreads();
        v16bf a  = frag_A(&As[mt * 16][0], 56, lane);
        v16bf b0 = frag_B(&BsT[nt0 * 16][0], 40, lane);
        v16bf b1 = frag_B(&BsT[(nt0 + 1) * 16][0], 40, lane);
        acc0 = wmma_bf16(a, b0, acc0);
        acc1 = wmma_bf16(a, b1, acc1);
        __syncthreads();
    }

    // spill v_d (bf16) into vt tile: C layout VGPR e -> M = e (+8 for hi half)
    {
        int m8 = (lane >> 4) * 8, n = lane & 15;
#pragma unroll
        for (int e = 0; e < 8; ++e) {
            int m = mt * 16 + e + m8;
            VT[m][nt0 * 16 + n]       = f2bf(acc0[e]);
            VT[m][(nt0 + 1) * 16 + n] = f2bf(acc1[e]);
        }
    }
    // time embedding cols 128..139, zero pad 140..159
#pragma unroll
    for (int e = 0; e < 4; ++e) {
        int idx = tid + e * 256;
        int m = idx >> 5, cc = idx & 31;
        float val = (cc < HT) ? (t[rb + m] * wtw[cc] + wtb[cc]) : 0.f;
        VT[m][HI + cc] = f2bf(val);
    }
    __syncthreads();

    // second GEMM entirely in LDS: inputs_tile = VT[32,160] @ W_vt^T[160,128]
    v8f o0 = {}, o1 = {};
#pragma unroll
    for (int kk = 0; kk < KPAD / 32; ++kk) {
        v16bf a  = frag_A(&VT[mt * 16][kk * 32], 168, lane);
        v16bf b0 = frag_B(&Wv[nt0 * 16][kk * 32], 168, lane);
        v16bf b1 = frag_B(&Wv[(nt0 + 1) * 16][kk * 32], 168, lane);
        o0 = wmma_bf16(a, b0, o0);
        o1 = wmma_bf16(a, b1, o1);
    }
    {
        int m8 = (lane >> 4) * 8, n = lane & 15;
#pragma unroll
        for (int e = 0; e < 8; ++e) {
            int row = rb + mt * 16 + e + m8;
            inputs[(size_t)row * HI + nt0 * 16 + n]       = f2bf(o0[e]);
            inputs[(size_t)row * HI + (nt0 + 1) * 16 + n] = f2bf(o1[e]);
        }
    }
}

// ---------------------------------------------------------------------------
// LSTM step s: gates[512,2048] = x_s @ W_ih^T + h @ W_hh^T + b ; cell update.
// Block tile: 32 batch rows x 64 H-cols, all 4 gates. 8 waves: mt=w>>2, nt=w&3.
// Each wave holds i/f/g/o accumulators for the SAME 16x16 coords -> pointwise
// cell update stays in registers. c updated in place (tile-exclusive).
// ---------------------------------------------------------------------------
__global__ __launch_bounds__(256) void k_lstm_step(
    const __bf16* __restrict__ inputs, // [8192, 128]
    const __bf16* __restrict__ WihP,   // panels [4][2048][32]
    const __bf16* __restrict__ WhhP,   // panels [16][2048][32]
    const float* __restrict__ b_ih,    // [2048]
    const float* __restrict__ b_hh,    // [2048]
    const __bf16* __restrict__ h_in,   // [512, 512]
    __bf16* __restrict__ h_out,        // [512, 512]
    float* __restrict__ c,             // [512, 512]
    int step)
{
    __shared__ __align__(16) __bf16 As[32][56];
    __shared__ __align__(16) __bf16 BsT[256][40];  // 4 gate segments x 64 cols

    const int tid  = threadIdx.x;
    const int lane = tid & 31;
    const int w    = tid >> 5;
    const int mt   = w >> 2;
    const int nt   = w & 3;
    const int rowBase = blockIdx.x * 32;
    const int colBase = blockIdx.y * 64;

    v8f acc[4] = {{}, {}, {}, {}};

    for (int phase = 0; phase < 2; ++phase) {
        const int K = phase ? HH : HI;
        const __bf16* WP = phase ? WhhP : WihP;
        for (int k0 = 0; k0 < K; k0 += 32) {
            if (tid < 128) {                    // A: 32x32 bf16, uint4 copies
                int m = tid >> 2, q = tid & 3;
                const __bf16* src = phase
                    ? (h_in   + (size_t)(rowBase + m) * HH + k0 + q * 8)
                    : (inputs + (size_t)((rowBase + m) * RW + step) * HI + k0 + q * 8);
                *(uint4*)&As[m][q * 8] = *(const uint4*)src;
            }
            {   // B: 256 cols x 32 k from panels, contiguous per column
                const __bf16* panel = WP + (size_t)(k0 >> 5) * (2048 * 32);
#pragma unroll
                for (int e = 0; e < 4; ++e) {
                    int idx = tid + e * 256;    // 1024 uint4
                    int col = idx >> 2, q = idx & 3;
                    int g = col >> 6, cc = col & 63;
                    *(uint4*)&BsT[col][q * 8] =
                        *(const uint4*)(panel + (size_t)(g * HH + colBase + cc) * 32 + q * 8);
                }
            }
            __syncthreads();
            v16bf a = frag_A(&As[mt * 16][0], 56, lane);
#pragma unroll
            for (int g = 0; g < 4; ++g) {
                v16bf b = frag_B(&BsT[g * 64 + nt * 16][0], 40, lane);
                acc[g] = wmma_bf16(a, b, acc[g]);
            }
            __syncthreads();
        }
    }

    // pointwise cell update (torch gate order: i, f, g, o)
    {
        int m8 = (lane >> 4) * 8, n = lane & 15;
#pragma unroll
        for (int e = 0; e < 8; ++e) {
            int row = rowBase + mt * 16 + e + m8;
            int col = colBase + nt * 16 + n;
            float gi = sigf (acc[0][e] + b_ih[col]          + b_hh[col]);
            float gf = sigf (acc[1][e] + b_ih[HH + col]     + b_hh[HH + col]);
            float gg = tanhf(acc[2][e] + b_ih[2 * HH + col] + b_hh[2 * HH + col]);
            float go = sigf (acc[3][e] + b_ih[3 * HH + col] + b_hh[3 * HH + col]);
            size_t idx = (size_t)row * HH + col;
            float cn = gf * c[idx] + gi * gg;
            c[idx] = cn;
            h_out[idx] = f2bf(go * tanhf(cn));
        }
    }
}

// ---------------------------------------------------------------------------
// pred[b] = h[b,:] . lin_w + lin_b  (one wave32 per row)
// ---------------------------------------------------------------------------
__global__ __launch_bounds__(256) void k_head(const __bf16* __restrict__ h,
                                              const float* __restrict__ lw,
                                              const float* __restrict__ lb,
                                              float* __restrict__ out) {
    int lane = threadIdx.x & 31;
    int row  = blockIdx.x * 8 + (threadIdx.x >> 5);
    float s = 0.f;
    for (int j = lane; j < HH; j += 32)
        s += bf2f(h[(size_t)row * HH + j]) * lw[j];
#pragma unroll
    for (int off = 16; off; off >>= 1)
        s += __shfl_xor(s, off, 32);
    if (lane == 0) out[row] = s + lb[0];
}

// ---------------------------------------------------------------------------
extern "C" void kernel_launch(void* const* d_in, const int* in_sizes, int n_in,
                              void* d_out, int out_size, void* d_ws, size_t ws_size,
                              hipStream_t stream) {
    (void)in_sizes; (void)n_in; (void)out_size; (void)ws_size;
    const float* v   = (const float*)d_in[0];
    const float* t   = (const float*)d_in[1];
    const float* Wd  = (const float*)d_in[2];
    const float* Wtw = (const float*)d_in[3];
    const float* Wtb = (const float*)d_in[4];
    const float* Wvt = (const float*)d_in[5];
    const float* Wih = (const float*)d_in[6];
    const float* Whh = (const float*)d_in[7];
    const float* bih = (const float*)d_in[8];
    const float* bhh = (const float*)d_in[9];
    const float* lw  = (const float*)d_in[10];
    const float* lb  = (const float*)d_in[11];
    float* out = (float*)d_out;

    char* ws = (char*)d_ws;
    size_t off = 0;
    auto wsalloc = [&](size_t bytes) -> void* {
        void* p = ws + off;
        off = (off + bytes + 255) & ~(size_t)255;
        return p;
    };
    __bf16* WdP    = (__bf16*)wsalloc((size_t)NN * HI * 2);        // panels [375][128][32]
    __bf16* WvtP   = (__bf16*)wsalloc((size_t)KPAD * HI * 2);      // panels [5][128][32]
    __bf16* WihP   = (__bf16*)wsalloc((size_t)HI * 4 * HH * 2);    // panels [4][2048][32]
    __bf16* WhhP   = (__bf16*)wsalloc((size_t)HH * 4 * HH * 2);    // panels [16][2048][32]
    __bf16* inputs = (__bf16*)wsalloc((size_t)ROWS * HI * 2);      // [8192,128]
    __bf16* hbuf0  = (__bf16*)wsalloc((size_t)BB * HH * 2);
    __bf16* hbuf1  = (__bf16*)wsalloc((size_t)BB * HH * 2);
    float*  cbuf   = (float*) wsalloc((size_t)BB * HH * 4);

    // weight prep (bf16, panelized [K/32][col][32] for wide coalesced copies)
    k_panel_cvt<<<(NN * HI + 255) / 256, 256, 0, stream>>>(Wd, WdP, 7, NN, NN);
    k_panel_cvt<<<(KPAD * HI + 255) / 256, 256, 0, stream>>>(Wvt, WvtP, 7, HI + HT, KPAD);
    k_panel_cvt<<<(HI * 4 * HH + 255) / 256, 256, 0, stream>>>(Wih, WihP, 11, HI, HI);
    k_panel_cvt<<<(HH * 4 * HH + 255) / 256, 256, 0, stream>>>(Whh, WhhP, 11, HH, HH);
    k_init_state<<<(BB * HH + 255) / 256, 256, 0, stream>>>(hbuf0, cbuf, BB * HH);

    // fused down-projection + time-embed + W_vt GEMM (streams 393MB of v)
    k_downproj_fused<<<ROWS / 32, 256, 0, stream>>>(v, t, Wtw, Wtb, WdP, WvtP, inputs);

    // 16 sequential LSTM steps; h ping-pongs, c in place
    for (int s = 0; s < RW; ++s) {
        __bf16* hin  = (s & 1) ? hbuf1 : hbuf0;
        __bf16* hout = (s & 1) ? hbuf0 : hbuf1;
        k_lstm_step<<<dim3(BB / 32, HH / 64), 256, 0, stream>>>(
            inputs, WihP, WhhP, bih, bhh, hin, hout, cbuf, s);
    }
    // step 15 (odd) wrote hbuf0
    k_head<<<BB / 8, 256, 0, stream>>>(hbuf0, lw, lb, out);
}
